// THOC_5995774345794
// MI455X (gfx1250) — matrix operand order repaired and checked by reading, not compiled, not run on hardware
//
#include <hip/hip_runtime.h>

typedef __attribute__((ext_vector_type(16))) __bf16 v16bf;
typedef __attribute__((ext_vector_type(8)))  __bf16 v8bf;
typedef __attribute__((ext_vector_type(8)))  float  v8f;

__device__ __forceinline__ __bf16 f2bf(float f) {
    unsigned u = __builtin_bit_cast(unsigned, f);
    unsigned r = u + 0x7FFFu + ((u >> 16) & 1u);   // round-to-nearest-even
    unsigned short h = (unsigned short)(r >> 16);
    return __builtin_bit_cast(__bf16, h);
}
__device__ __forceinline__ float bf2f(__bf16 b) {
    unsigned short us = __builtin_bit_cast(unsigned short, b);
    return __builtin_bit_cast(float, ((unsigned)us) << 16);
}

// ---------- WMMA fragment loaders (contiguous k order: K = kc + 16*half + e, same in A and B) ----------
// A 16x32 bf16 frag from bf16 memory. prow = &A[row][kc]. element e: K = kc + 16*half + e
__device__ __forceinline__ v16bf load_a_bf16(const __bf16* prow, int half) {
    const __bf16* p = prow + 16 * half;
    v8bf lo = *(const v8bf*)p;          // K offsets 16*half + 0..7
    v8bf hi = *(const v8bf*)(p + 8);    // K offsets 16*half + 8..15
    v16bf a;
#pragma unroll
    for (int e = 0; e < 8; ++e) { a[e] = lo[e]; a[8 + e] = hi[e]; }
    return a;
}

// B 32x16 bf16 frag from bf16 W[Nc][Ka] row-major (A @ W^T). lane: N=lane&15, K = kc + 16*half + e
__device__ __forceinline__ v16bf load_b_bf16(const __bf16* W, int ldw, int col0, int kbase, int lane) {
    return *(const v16bf*)(W + (size_t)(col0 + (lane & 15)) * ldw + kbase + 16 * (lane >> 4));
}

// ---------- generic bf16 WMMA GEMM: act(A[M x Ka] @ W[Nc x Ka]^T + bias) -> C (f32) and/or Cb (bf16)
// concat mode (A2 != nullptr): row = kl*cN + t; k < Hd reads A (=f_hat[t][kl][k]); k >= Hd reads A2[t][k-Hd]
__global__ __launch_bounds__(256) void wmma_gemm_kernel(
    const __bf16* __restrict__ A, const __bf16* __restrict__ A2,
    const __bf16* __restrict__ W, const float* __restrict__ bias,
    float* __restrict__ C, __bf16* __restrict__ Cb,
    int Nc, int Ka, int lda, int act, int cN, int cKl, int Hd)
{
    const int lane = threadIdx.x & 31;
    const int wave = threadIdx.x >> 5;
    const int nTile = blockIdx.x * 8 + wave;
    if (nTile * 16 >= Nc) return;                 // wave-uniform exit, EXEC stays full per wave
    const int m = lane & 15, half = lane >> 4;
    const int row = blockIdx.y * 16 + m;

    const __bf16* aBase;
    const __bf16* a2Base = nullptr;
    if (A2) {
        int t = row % cN, kl = row / cN;
        aBase  = A + ((size_t)t * cKl + kl) * (size_t)Hd;
        a2Base = A2 + (size_t)t * Hd;
    } else {
        aBase = A + (size_t)row * lda;
    }

    v8f acc = {};
    for (int kc = 0; kc < Ka; kc += 32) {
        const __bf16* prow = (!A2 || kc < Hd) ? (aBase + kc) : (a2Base + (kc - Hd));
        if (kc + 32 < Ka) __builtin_prefetch(prow + 32, 0, 0);   // global_prefetch_b8
        v16bf a = load_a_bf16(prow, half);
        v16bf b = load_b_bf16(W, Ka, nTile * 16, kc, lane);
        acc = __builtin_amdgcn_wmma_f32_16x16x32_bf16(false, a, false, b, (short)0, acc, false, false);
    }

    const int n = nTile * 16 + m;
    const float bv = bias ? bias[n] : 0.f;
    const size_t base = (size_t)(blockIdx.y * 16) * Nc + n;
#pragma unroll
    for (int r = 0; r < 8; ++r) {                 // C frag: M = r + 8*half, N = lane&15
        float v = acc[r] + bv;
        if (act) v = fmaxf(v, 0.f);
        size_t idx = base + (size_t)(r + 8 * half) * Nc;
        if (C)  C[idx]  = v;
        if (Cb) Cb[idx] = f2bf(v);
    }
}

// ---------- f32 -> bf16 convert (weights / x, one pass each) ----------
__global__ __launch_bounds__(256) void f32_to_bf16_kernel(const float* __restrict__ s,
                                                          __bf16* __restrict__ d, int n)
{
    int i = blockIdx.x * 256 + threadIdx.x;
    if (i < n) d[i] = f2bf(s[i]);
}

// ---------- persistent GRU scan: one workgroup, 16 waves, WMMA recurrent GEMM each step ----------
// gi [S][Bt][768] f32 (includes b_ih), Whh bf16 [768][256], bhh [768], out bf16 [S][Bt][256]
// LDS: h f32 [32][256] (32KB) | h bf16 in WMMA-A layout [2][8*512] (16KB) | gh f32 [32][768] (96KB)
template <int MT>
__global__ __launch_bounds__(512) void gru_scan_kernel(
    const float* __restrict__ gi, const __bf16* __restrict__ Whh,
    const float* __restrict__ bhh, __bf16* __restrict__ out, int S, int Bt)
{
    extern __shared__ char smem_raw[];
    float*  h  = (float*)smem_raw;                  // 32*256 f32
    __bf16* hb = (__bf16*)(smem_raw + 32768);       // 32*256 bf16, A-fragment swizzled
    float*  gh = (float*)(smem_raw + 49152);        // 32*768 f32

    const int lane = threadIdx.x & 31;
    const int wave = threadIdx.x >> 5;              // 0..15
    const int m = lane & 15, half = lane >> 4;

    for (int i = threadIdx.x; i < 32 * 256; i += 512) { h[i] = 0.f; hb[i] = f2bf(0.f); }
    __syncthreads();

    for (int s = 0; s < S; ++s) {
        // --- gh = h @ Whh^T : each wave owns 3 N-tiles (48 tiles = 768 cols), MT M-tiles ---
        v8f acc[MT][3];
#pragma unroll
        for (int mt = 0; mt < MT; ++mt)
#pragma unroll
            for (int t2 = 0; t2 < 3; ++t2) acc[mt][t2] = (v8f){};

#pragma unroll
        for (int kc8 = 0; kc8 < 8; ++kc8) {
            const int kc = kc8 * 32;
            v16bf afrag[MT];
#pragma unroll
            for (int mt = 0; mt < MT; ++mt)
                afrag[mt] = *(const v16bf*)(hb + mt * 4096 + kc8 * 512 + (half * 16 + m) * 16);
#pragma unroll
            for (int t2 = 0; t2 < 3; ++t2) {
                const int nTile = wave * 3 + t2;
                v16bf bfrag = *(const v16bf*)(Whh + (size_t)(nTile * 16 + m) * 256 + kc + 16 * half);
#pragma unroll
                for (int mt = 0; mt < MT; ++mt)
                    acc[mt][t2] = __builtin_amdgcn_wmma_f32_16x16x32_bf16(
                        false, afrag[mt], false, bfrag, (short)0, acc[mt][t2], false, false);
            }
        }
#pragma unroll
        for (int mt = 0; mt < MT; ++mt)
#pragma unroll
            for (int t2 = 0; t2 < 3; ++t2) {
                const int n = (wave * 3 + t2) * 16 + m;
#pragma unroll
                for (int r = 0; r < 8; ++r)
                    gh[(mt * 16 + r + 8 * half) * 768 + n] = acc[mt][t2][r];
            }
        __syncthreads();

        // --- fused gate update: r,z,n gates; write h (f32 + swizzled bf16) and stream out (bf16) ---
        const int total = Bt << 8;
        for (int idx = threadIdx.x; idx < total; idx += 512) {
            const int b = idx >> 8, j = idx & 255;
            const float* gr = gi + ((size_t)s * Bt + b) * 768;
            float ir = gr[j], iz = gr[256 + j], inn = gr[512 + j];
            float hr = gh[b * 768 + j] + bhh[j];
            float hz = gh[b * 768 + 256 + j] + bhh[256 + j];
            float hn = gh[b * 768 + 512 + j] + bhh[512 + j];
            float rg = 1.f / (1.f + __expf(-(ir + hr)));
            float zg = 1.f / (1.f + __expf(-(iz + hz)));
            float ng = tanhf(inn + rg * hn);
            float hv = (1.f - zg) * ng + zg * h[(b << 8) + j];
            h[(b << 8) + j] = hv;
            __bf16 hvb = f2bf(hv);
            const int kc8 = j >> 5, rr = j & 31;
            const int e  = rr & 15;
            const int hk = rr >> 4;
            hb[(b >> 4) * 4096 + kc8 * 512 + (hk * 16 + (b & 15)) * 16 + e] = hvb;
            out[((size_t)s * Bt + b) * 256 + j] = hvb;
        }
        __syncthreads();
    }
}

// ---------- cosine similarity + softmax: P[row][k] = softmax_k( cos(F[row], C[k]) ) ----------
template <int KL>
__global__ __launch_bounds__(256) void cos_softmax_kernel(
    const __bf16* __restrict__ F, const float* __restrict__ Cent,
    float* __restrict__ P, int rows)
{
    __shared__ float cn[KL * 256];
    __shared__ float cnorm[KL];
    if (threadIdx.x < KL) {
        float s2 = 0.f;
        for (int hh = 0; hh < 256; ++hh) { float v = Cent[threadIdx.x * 256 + hh]; s2 += v * v; }
        cnorm[threadIdx.x] = fmaxf(sqrtf(s2), 1e-8f);
    }
    __syncthreads();
    for (int i = threadIdx.x; i < KL * 256; i += 256) cn[i] = Cent[i] / cnorm[i >> 8];
    __syncthreads();

    int row = blockIdx.x * 256 + threadIdx.x;
    if (row >= rows) return;
    const __bf16* f = F + (size_t)row * 256;
    float dot[KL];
#pragma unroll
    for (int k = 0; k < KL; ++k) dot[k] = 0.f;
    float nrm = 0.f;
    for (int hh = 0; hh < 256; ++hh) {
        float v = bf2f(f[hh]); nrm += v * v;
#pragma unroll
        for (int k = 0; k < KL; ++k) dot[k] += v * cn[k * 256 + hh];
    }
    float inv_n = 1.f / fmaxf(sqrtf(nrm), 1e-8f);
    float mx = -1e30f;
#pragma unroll
    for (int k = 0; k < KL; ++k) { dot[k] *= inv_n; mx = fmaxf(mx, dot[k]); }
    float s = 0.f;
#pragma unroll
    for (int k = 0; k < KL; ++k) { dot[k] = __expf(dot[k] - mx); s += dot[k]; }
    float inv = 1.f / s;
    float* pp = P + (size_t)row * KL;
#pragma unroll
    for (int k = 0; k < KL; ++k) pp[k] = dot[k] * inv;
}

// ---------- R propagation: Rnew[t][k] = norm_k( sum_c P[c][t][k] * Rold[t][c] ) ----------
template <int KP, int KL>
__global__ __launch_bounds__(256) void r_update_kernel(
    const float* __restrict__ P, const float* __restrict__ Rold,
    float* __restrict__ Rnew, int N)
{
    int t = blockIdx.x * 256 + threadIdx.x;
    if (t >= N) return;
    float acc[KL];
#pragma unroll
    for (int k = 0; k < KL; ++k) acc[k] = 0.f;
    for (int c = 0; c < KP; ++c) {
        float rc = Rold[(size_t)t * KP + c];
        const float* pr = P + ((size_t)c * N + t) * KL;
#pragma unroll
        for (int k = 0; k < KL; ++k) acc[k] += pr[k] * rc;
    }
    float s = 0.f;
#pragma unroll
    for (int k = 0; k < KL; ++k) s += acc[k];
    float inv = 1.f / s;
#pragma unroll
    for (int k = 0; k < KL; ++k) Rnew[(size_t)t * KL + k] = acc[k] * inv;
}

// ---------- f_hat[t][j][h] = sum_k P[k][t][j] * Y[k][t][h]  (bf16 out) ----------
template <int KP>
__global__ __launch_bounds__(256) void fhat_kernel(
    const float* __restrict__ P, const float* __restrict__ Y,
    __bf16* __restrict__ FH, int N, int KL)
{
    size_t idx = (size_t)blockIdx.x * 256 + threadIdx.x;
    int hcol = (int)(idx & 255);
    size_t tj = idx >> 8;
    int j = (int)(tj % KL);
    size_t t = tj / KL;
    if (t >= (size_t)N) return;
    float s = 0.f;
#pragma unroll
    for (int k = 0; k < KP; ++k)
        s += P[((size_t)k * N + t) * KL + j] * Y[((size_t)k * N + t) * 256 + hcol];
    FH[idx] = f2bf(s);
}

// ---------- anomaly[t] = sum_f R[t][f] * sum_c (1 - cos(FH[t][f], C[c])) ----------
__global__ __launch_bounds__(256) void anomaly_kernel(
    const __bf16* __restrict__ FH, const float* __restrict__ Cc,
    const float* __restrict__ R, float* __restrict__ outp, int N)
{
    __shared__ float cn[6 * 256];
    __shared__ float cnorm[6];
    if (threadIdx.x < 6) {
        float s2 = 0.f;
        for (int hh = 0; hh < 256; ++hh) { float v = Cc[threadIdx.x * 256 + hh]; s2 += v * v; }
        cnorm[threadIdx.x] = fmaxf(sqrtf(s2), 1e-8f);
    }
    __syncthreads();
    for (int i = threadIdx.x; i < 6 * 256; i += 256) cn[i] = Cc[i] / cnorm[i >> 8];
    __syncthreads();

    int t = blockIdx.x * 256 + threadIdx.x;
    if (t >= N) return;
    float a = 0.f;
    for (int f = 0; f < 6; ++f) {
        const __bf16* fr = FH + ((size_t)t * 6 + f) * 256;
        float dot[6] = {0, 0, 0, 0, 0, 0};
        float nrm = 0.f;
        for (int hh = 0; hh < 256; ++hh) {
            float v = bf2f(fr[hh]); nrm += v * v;
#pragma unroll
            for (int c2 = 0; c2 < 6; ++c2) dot[c2] += v * cn[c2 * 256 + hh];
        }
        float inv_n = 1.f / fmaxf(sqrtf(nrm), 1e-8f);
        float ssum = 0.f;
#pragma unroll
        for (int c2 = 0; c2 < 6; ++c2) ssum += dot[c2] * inv_n;
        a += R[(size_t)t * 6 + f] * (6.f - ssum);
    }
    outp[t] = a;
}

// =====================================================================================
extern "C" void kernel_launch(void* const* d_in, const int* in_sizes, int n_in,
                              void* d_out, int out_size, void* d_ws, size_t ws_size,
                              hipStream_t stream)
{
    (void)in_sizes; (void)n_in; (void)out_size; (void)ws_size;
    const float* x        = (const float*)d_in[0];
    const float* W_ih[3]  = {(const float*)d_in[1], (const float*)d_in[5],  (const float*)d_in[9]};
    const float* W_hh[3]  = {(const float*)d_in[2], (const float*)d_in[6],  (const float*)d_in[10]};
    const float* b_ih[3]  = {(const float*)d_in[3], (const float*)d_in[7],  (const float*)d_in[11]};
    const float* b_hh[3]  = {(const float*)d_in[4], (const float*)d_in[8],  (const float*)d_in[12]};
    const float* W_lin    = (const float*)d_in[13];
    const float* b_lin    = (const float*)d_in[14];
    const float* W_mlp    = (const float*)d_in[15];
    const float* b_mlp    = (const float*)d_in[16];
    const float* Cn[3]    = {(const float*)d_in[17], (const float*)d_in[18], (const float*)d_in[19]};
    float* outp = (float*)d_out;

    constexpr int T = 2048, B = 8, IN = 64, H = 256;
    constexpr int N = T * B;                       // 16384

    // ---- workspace carve-out (bf16-native activation flow) ----
    char* ws = (char*)d_ws;
    size_t off = 0;
    auto alloc = [&](size_t bytes) -> void* {
        off = (off + 255) & ~(size_t)255;
        void* p = ws + off; off += bytes; return p;
    };
    __bf16* xb    = (__bf16*)alloc((size_t)N * IN * 2);
    __bf16* outsb[3];
    for (int i = 0; i < 3; ++i) outsb[i] = (__bf16*)alloc((size_t)N * H * 2);
    float*  gi    = (float*)alloc((size_t)N * 768 * 4);
    __bf16* wihb  = (__bf16*)alloc((size_t)768 * H * 2);     // per-layer, reused
    __bf16* whhb  = (__bf16*)alloc((size_t)768 * H * 2);
    __bf16* wlinb = (__bf16*)alloc((size_t)H * H * 2);
    __bf16* wmlpb = (__bf16*)alloc((size_t)H * 2 * H * 2);
    float*  P     = (float*)alloc((size_t)18 * N * 18 * 4);
    float*  Ra    = (float*)alloc((size_t)N * 18 * 4);
    float*  Rb    = (float*)alloc((size_t)N * 18 * 4);
    __bf16* fbarb = (__bf16*)alloc((size_t)18 * N * H * 2);
    float*  ybuf  = (float*)alloc((size_t)18 * N * H * 4);
    __bf16* fhatb = (__bf16*)alloc((size_t)N * 18 * H * 2);

    const int SCAN_SMEM = 32 * 256 * 4 + 32 * 256 * 2 + 32 * 768 * 4;   // 147456 B < 320KB WGP LDS
    (void)hipFuncSetAttribute(reinterpret_cast<const void*>(gru_scan_kernel<1>),
                              hipFuncAttributeMaxDynamicSharedMemorySize, SCAN_SMEM);
    (void)hipFuncSetAttribute(reinterpret_cast<const void*>(gru_scan_kernel<2>),
                              hipFuncAttributeMaxDynamicSharedMemorySize, SCAN_SMEM);

    // ---- one-time f32 -> bf16 conversions ----
    f32_to_bf16_kernel<<<(N * IN + 255) / 256, 256, 0, stream>>>(x, xb, N * IN);
    f32_to_bf16_kernel<<<(H * H + 255) / 256, 256, 0, stream>>>(W_lin, wlinb, H * H);
    f32_to_bf16_kernel<<<(H * 2 * H + 255) / 256, 256, 0, stream>>>(W_mlp, wmlpb, H * 2 * H);

    // ======== dilated GRU stack (reshapes are no-ops on contiguous [T*B, C]) ========
    const int ins[3] = {IN, H, H};
    const __bf16* act_in = xb;
    for (int i = 0; i < 3; ++i) {
        int Ka = ins[i];
        f32_to_bf16_kernel<<<(768 * Ka + 255) / 256, 256, 0, stream>>>(W_ih[i], wihb, 768 * Ka);
        f32_to_bf16_kernel<<<(768 * H + 255) / 256, 256, 0, stream>>>(W_hh[i], whhb, 768 * H);
        dim3 g(6, N / 16);  // 768 cols = 48 tiles = 6 blocks x 8 waves
        wmma_gemm_kernel<<<g, 256, 0, stream>>>(act_in, nullptr, wihb, b_ih[i], gi, nullptr,
                                                768, Ka, Ka, 0, 0, 0, 0);
        int rate = 1 << i, Bt = rate * B, S = T / rate;
        if (Bt <= 16)
            gru_scan_kernel<1><<<1, 512, SCAN_SMEM, stream>>>(gi, whhb, b_hh[i], outsb[i], S, Bt);
        else
            gru_scan_kernel<2><<<1, 512, SCAN_SMEM, stream>>>(gi, whhb, b_hh[i], outsb[i], S, Bt);
        act_in = outsb[i];
    }

    // ======== hierarchical soft clustering ========
    // ---- l = 0: f_bar = outs[0] (Kprev=1, Kl=18) ----
    cos_softmax_kernel<18><<<N / 256, 256, 0, stream>>>(outsb[0], Cn[0], P, N);
    hipMemcpyAsync(Ra, P, (size_t)N * 18 * 4, hipMemcpyDeviceToDevice, stream);   // R = P[0]
    { dim3 g(2, N / 16);
      wmma_gemm_kernel<<<g, 256, 0, stream>>>(outsb[0], nullptr, wlinb, b_lin, ybuf, nullptr,
                                              H, H, H, 1, 0, 0, 0); }
    fhat_kernel<1><<<(unsigned)((size_t)N * 18 * H / 256), 256, 0, stream>>>(P, ybuf, fhatb, N, 18);
    { dim3 g(2, 18 * N / 16);   // MLP on concat(f_hat, outs[1]) : K = 512, bf16 output
      wmma_gemm_kernel<<<g, 256, 0, stream>>>(fhatb, outsb[1], wmlpb, b_mlp, nullptr, fbarb,
                                              H, 2 * H, 0, 0, N, 18, H); }

    // ---- l = 1: Kprev=18, Kl=12 ----
    cos_softmax_kernel<12><<<18 * N / 256, 256, 0, stream>>>(fbarb, Cn[1], P, 18 * N);
    r_update_kernel<18, 12><<<N / 256, 256, 0, stream>>>(P, Ra, Rb, N);
    { dim3 g(2, 18 * N / 16);
      wmma_gemm_kernel<<<g, 256, 0, stream>>>(fbarb, nullptr, wlinb, b_lin, ybuf, nullptr,
                                              H, H, H, 1, 0, 0, 0); }
    fhat_kernel<18><<<(unsigned)((size_t)N * 12 * H / 256), 256, 0, stream>>>(P, ybuf, fhatb, N, 12);
    { dim3 g(2, 12 * N / 16);
      wmma_gemm_kernel<<<g, 256, 0, stream>>>(fhatb, outsb[2], wmlpb, b_mlp, nullptr, fbarb,
                                              H, 2 * H, 0, 0, N, 12, H); }

    // ---- l = 2: Kprev=12, Kl=6 ----
    cos_softmax_kernel<6><<<12 * N / 256, 256, 0, stream>>>(fbarb, Cn[2], P, 12 * N);
    r_update_kernel<12, 6><<<N / 256, 256, 0, stream>>>(P, Rb, Ra, N);
    { dim3 g(2, 12 * N / 16);
      wmma_gemm_kernel<<<g, 256, 0, stream>>>(fbarb, nullptr, wlinb, b_lin, ybuf, nullptr,
                                              H, H, H, 1, 0, 0, 0); }
    fhat_kernel<12><<<(unsigned)((size_t)N * 6 * H / 256), 256, 0, stream>>>(P, ybuf, fhatb, N, 6);

    // ---- anomaly score ----
    anomaly_kernel<<<N / 256, 256, 0, stream>>>(fhatb, Cn[2], Ra, outp, N);
}
